// GroupQLinear_17678085390337
// MI455X (gfx1250) — compile-verified, matrix operand order checked by reading, not run on hardware
//
#include <hip/hip_runtime.h>
#include <hip/hip_bf16.h>

typedef _Float16 v4h  __attribute__((ext_vector_type(4)));
typedef _Float16 v8h  __attribute__((ext_vector_type(8)));
typedef _Float16 v16h __attribute__((ext_vector_type(16)));
typedef float    v8f  __attribute__((ext_vector_type(8)));
typedef int      v4i  __attribute__((ext_vector_type(4)));

#define HDIM   4096          // H == K == O for this problem
#define GSZ    32            // RESIDUAL_GROUP
#define NG     128           // groups per token
#define RSTEPS 64

#define AS1 __attribute__((address_space(1)))
#define AS3 __attribute__((address_space(3)))

// ---------------------------------------------------------------------------
// CDNA5 async global->LDS copy (ASYNCcnt-tracked), 16B per lane.
// ---------------------------------------------------------------------------
__device__ __forceinline__ void async_cp_b128(const void* gsrc, void* ldst) {
#if __has_builtin(__builtin_amdgcn_global_load_async_to_lds_b128)
    __builtin_amdgcn_global_load_async_to_lds_b128(
        (AS1 v4i*)gsrc, (AS3 v4i*)ldst, 0, 0);
#else
    unsigned lds = (unsigned)(size_t)((AS3 void*)ldst);
    asm volatile("global_load_async_to_lds_b128 %0, %1, off"
                 :: "v"(lds), "v"(gsrc) : "memory");
#endif
}

__device__ __forceinline__ void wait_async0() {
#if __has_builtin(__builtin_amdgcn_s_wait_asynccnt)
    __builtin_amdgcn_s_wait_asynccnt(0);
#else
    asm volatile("s_wait_asynccnt 0x0" ::: "memory");
#endif
}

// ---------------------------------------------------------------------------
// Kernel 1: per-token group-wise quantization -> dequantized f16 activations
// One 128-thread block per token; thread g owns group g (32 floats).
// ---------------------------------------------------------------------------
__global__ __launch_bounds__(128) void quant_kernel(const float* __restrict__ x,
                                                    _Float16* __restrict__ aq) {
    const int m = blockIdx.x;
    const int g = threadIdx.x;              // 0..127
    const float* row = x + (size_t)m * HDIM + g * GSZ;

    float v[GSZ];
    const float4* p = (const float4*)row;
#pragma unroll
    for (int i = 0; i < 8; ++i) {
        float4 t = p[i];
        v[4*i+0] = t.x; v[4*i+1] = t.y; v[4*i+2] = t.z; v[4*i+3] = t.w;
    }

    float amax = 0.0f;
#pragma unroll
    for (int i = 0; i < GSZ; ++i) amax = fmaxf(amax, fabsf(v[i]));
    const float delta = fmaxf(amax, 1e-5f) * (1.0f / 127.0f);

    __shared__ float sDelta[NG];
    __shared__ float sRed[NG];
    __shared__ float sE[NG];
    __shared__ float sLoss[RSTEPS];
    __shared__ int   sIdx[RSTEPS];

    sDelta[g] = delta;
    sRed[g]   = delta;
    __syncthreads();

    // max-reduce -> delta_base
#pragma unroll
    for (int s = 64; s > 0; s >>= 1) {
        if (g < s) sRed[g] = fmaxf(sRed[g], sRed[g + s]);
        __syncthreads();
    }
    const float dbase = sRed[0];
    __syncthreads();

    // sum-reduce |dbase - delta|  (dbase >= delta always)
    sRed[g] = dbase - delta;
    __syncthreads();
#pragma unroll
    for (int s = 64; s > 0; s >>= 1) {
        if (g < s) sRed[g] += sRed[g + s];
        __syncthreads();
    }
    const float Rinit = fmaxf(sRed[0] * (1.0f / NG) * 0.25f, 1e-8f);

    float e = floorf((delta - dbase) / Rinit);
    e = fminf(fmaxf(e, -7.0f), 0.0f);
    sE[g] = e;
    __syncthreads();

    // 64 candidate residual steps; thread i evaluates candidate i
    if (g < RSTEPS) {
        const float Rc = dbase * ((float)g * (1.0f / (RSTEPS - 1)));
        float acc = 0.0f;
#pragma unroll 4
        for (int j = 0; j < NG; ++j) {
            float d = sDelta[j] - dbase - sE[j] * Rc;
            acc = fmaf(d, d, acc);
        }
        sLoss[g] = acc * (1.0f / NG);
        sIdx[g]  = g;
    }
    __syncthreads();

    // argmin with first-min tie rule (ties keep the lower/left index)
#pragma unroll
    for (int s = 32; s > 0; s >>= 1) {
        if (g < s) {
            float lo = sLoss[g], hi = sLoss[g + s];
            if (hi < lo) { sLoss[g] = hi; sIdx[g] = sIdx[g + s]; }
        }
        __syncthreads();
    }
    const float bestR = dbase * ((float)sIdx[0] * (1.0f / (RSTEPS - 1)));
    const float drec  = fmaxf(dbase + e * bestR, 1e-5f);
    const float inv   = 1.0f / drec;

    _Float16* dst = aq + (size_t)m * HDIM + g * GSZ;
#pragma unroll
    for (int c = 0; c < 4; ++c) {
        v8h h;
#pragma unroll
        for (int i = 0; i < 8; ++i)
            h[i] = (_Float16)(rintf(v[c * 8 + i] * inv) * drec);
        *(v8h*)(dst + c * 8) = h;
    }
}

// ---------------------------------------------------------------------------
// Kernel 1b: one-time fp32 -> f16 weight conversion (stays resident in L2)
// ---------------------------------------------------------------------------
__global__ __launch_bounds__(256) void wconv_kernel(const float* __restrict__ W,
                                                    _Float16* __restrict__ W16) {
    const size_t i = ((size_t)blockIdx.x * 256 + threadIdx.x) * 8;
    float4 a = *(const float4*)(W + i);
    float4 b = *(const float4*)(W + i + 4);
    v8h h = { (_Float16)a.x, (_Float16)a.y, (_Float16)a.z, (_Float16)a.w,
              (_Float16)b.x, (_Float16)b.y, (_Float16)b.z, (_Float16)b.w };
    *(v8h*)(W16 + i) = h;
}

// ---------------------------------------------------------------------------
// Kernel 2: y = Aq(f16) @ W^T(f16) + bias via v_wmma_f32_16x16x32_f16
// 128x128 workgroup tile, 256 threads = 8 waves (2 in M x 4 in N),
// each wave owns 64x32 = 4x2 WMMA accumulators. K-chunk = 32.
// W16PATH: tiles filled with GLOBAL_LOAD_ASYNC_TO_LDS_B128 (ASYNCcnt),
// else: fp32 weight converted in-kernel (register-staged fills).
// ---------------------------------------------------------------------------
#define TILE_M 128
#define TILE_N 128
#define TILE_K 32
#define LDSS   40            // f16 elements per LDS row (32 + 8 pad)

__device__ __forceinline__ v16h ldfrag(const _Float16* base) {
    v8h lo = *(const v8h*)(base);
    v8h hi = *(const v8h*)(base + 16);
    return __builtin_shufflevector(lo, hi, 0,1,2,3,4,5,6,7,8,9,10,11,12,13,14,15);
}

template <bool W16PATH>
__global__ __launch_bounds__(256) void gemm_kernel(const _Float16* __restrict__ A,
                                                   const _Float16* __restrict__ W16,
                                                   const float* __restrict__ Wf,
                                                   const float* __restrict__ bias,
                                                   float* __restrict__ out) {
    __shared__ _Float16 As[2][TILE_M * LDSS];
    __shared__ _Float16 Bs[2][TILE_N * LDSS];

    const int n0   = blockIdx.x * TILE_N;
    const int m0   = blockIdx.y * TILE_M;
    const int tid  = threadIdx.x;
    const int wid  = tid >> 5;
    const int lane = tid & 31;
    const int waveM = wid >> 2;          // 0..1
    const int waveN = wid & 3;           // 0..3
    const int lr    = lane & 15;         // row within a 16-wide fragment
    const int kb    = (lane >> 4) * 8;   // K sub-offset per documented layout

    v8f acc[4][2] = {};

    // ---- async f16 tile fill: 512 x 16B chunks, 2 per thread ----------
    auto issueTile = [&](const _Float16* base, int r0, int k0, _Float16* lds) {
#pragma unroll
        for (int i = 0; i < 2; ++i) {
            int c   = tid + i * 256;
            int row = c >> 2;
            int ko  = (c & 3) * 8;
            async_cp_b128(base + (size_t)(r0 + row) * HDIM + k0 + ko,
                          lds + row * LDSS + ko);
        }
    };

    // ---- fallback register-staged fills -------------------------------
    v8h    ra[2];
    float4 rb[4];
    auto loadA = [&](int k0) {
#pragma unroll
        for (int i = 0; i < 2; ++i) {
            int c = tid + i * 256, row = c >> 2, ko = (c & 3) * 8;
            ra[i] = *(const v8h*)(A + (size_t)(m0 + row) * HDIM + k0 + ko);
        }
    };
    auto storeA = [&](int buf) {
#pragma unroll
        for (int i = 0; i < 2; ++i) {
            int c = tid + i * 256, row = c >> 2, ko = (c & 3) * 8;
            *(v8h*)(&As[buf][row * LDSS + ko]) = ra[i];
        }
    };
    auto loadB = [&](int k0) {
#pragma unroll
        for (int i = 0; i < 4; ++i) {
            int c = tid + i * 256, row = c >> 3, ko = (c & 7) * 4;
            rb[i] = *(const float4*)(Wf + (size_t)(n0 + row) * HDIM + k0 + ko);
        }
    };
    auto storeB = [&](int buf) {
#pragma unroll
        for (int i = 0; i < 4; ++i) {
            int c = tid + i * 256, row = c >> 3, ko = (c & 7) * 4;
            v4h h = { (_Float16)rb[i].x, (_Float16)rb[i].y,
                      (_Float16)rb[i].z, (_Float16)rb[i].w };
            *(v4h*)(&Bs[buf][row * LDSS + ko]) = h;
        }
    };

    // ---- prologue -----------------------------------------------------
    if constexpr (W16PATH) {
        issueTile(A,   m0, 0, As[0]);
        issueTile(W16, n0, 0, Bs[0]);
        wait_async0();
    } else {
        loadA(0); loadB(0);
        storeA(0); storeB(0);
    }
    __syncthreads();

    const int KSTEPS = HDIM / TILE_K;    // 128

    for (int kc = 0; kc < KSTEPS; ++kc) {
        const int cur = kc & 1;

        if (kc + 1 < KSTEPS) {
            if constexpr (W16PATH) {
                issueTile(A,   m0, (kc + 1) * TILE_K, As[cur ^ 1]);
                issueTile(W16, n0, (kc + 1) * TILE_K, Bs[cur ^ 1]);
            } else {
                loadA((kc + 1) * TILE_K);
                loadB((kc + 1) * TILE_K);
            }
        }
        if (kc + 2 < KSTEPS) {           // warm L2/WGP$ two steps ahead
            __builtin_prefetch(A + (size_t)(m0 + (tid >> 1)) * HDIM + (kc + 2) * TILE_K, 0, 3);
            if constexpr (W16PATH)
                __builtin_prefetch(W16 + (size_t)(n0 + (tid >> 1)) * HDIM + (kc + 2) * TILE_K, 0, 3);
            else
                __builtin_prefetch(Wf + (size_t)(n0 + (tid >> 1)) * HDIM + (kc + 2) * TILE_K, 0, 3);
        }

        // ---- fragments + 8 WMMAs per wave -----------------------------
        v16h af[4], bf[2];
#pragma unroll
        for (int i = 0; i < 4; ++i)
            af[i] = ldfrag(&As[cur][(waveM * 64 + i * 16 + lr) * LDSS + kb]);
#pragma unroll
        for (int j = 0; j < 2; ++j)
            bf[j] = ldfrag(&Bs[cur][(waveN * 32 + j * 16 + lr) * LDSS + kb]);

#pragma unroll
        for (int i = 0; i < 4; ++i)
#pragma unroll
            for (int j = 0; j < 2; ++j)
                acc[i][j] = __builtin_amdgcn_wmma_f32_16x16x32_f16(
                    false, af[i], false, bf[j],
                    (short)0, acc[i][j], false, false);

        if (kc + 1 < KSTEPS) {
            if constexpr (W16PATH) {
                wait_async0();           // this wave's LDS fills landed
            } else {
                storeA(cur ^ 1);
                storeB(cur ^ 1);
            }
        }
        __syncthreads();                 // all waves' fills/reads ordered
    }

    // ---- epilogue: D layout — VGPR r: lanes0-15 M=r, lanes16-31 M=r+8 ----
#pragma unroll
    for (int j = 0; j < 2; ++j) {
        const int col = n0 + waveN * 32 + j * 16 + lr;
        const float b = bias[col];
#pragma unroll
        for (int i = 0; i < 4; ++i) {
            const int rbase = m0 + waveM * 64 + i * 16 + ((lane >> 4) * 8);
#pragma unroll
            for (int r = 0; r < 8; ++r)
                out[(size_t)(rbase + r) * HDIM + col] = acc[i][j][r] + b;
        }
    }
}

// ---------------------------------------------------------------------------
extern "C" void kernel_launch(void* const* d_in, const int* in_sizes, int n_in,
                              void* d_out, int out_size, void* d_ws, size_t ws_size,
                              hipStream_t stream) {
    const float* x    = (const float*)d_in[0];   // [B,T,H] f32
    const float* w    = (const float*)d_in[1];   // [O,H]   f32
    const float* bias = (const float*)d_in[2];   // [O]     f32
    float*       out  = (float*)d_out;           // [B,T,O] f32

    const int M = in_sizes[0] / HDIM;            // B*T = 8192

    const size_t A_BYTES = (size_t)M * HDIM * sizeof(_Float16);        // 64 MB
    const size_t W_BYTES = (size_t)HDIM * HDIM * sizeof(_Float16);     // 32 MB

    _Float16* aq  = (_Float16*)d_ws;
    _Float16* w16 = (_Float16*)((char*)d_ws + A_BYTES);
    const bool w16path = ws_size >= A_BYTES + W_BYTES;

    quant_kernel<<<M, 128, 0, stream>>>(x, aq);

    dim3 grid(HDIM / TILE_N, M / TILE_M);        // (32, 64)
    if (w16path) {
        wconv_kernel<<<(int)((size_t)HDIM * HDIM / (256 * 8)), 256, 0, stream>>>(w, w16);
        gemm_kernel<true><<<grid, 256, 0, stream>>>(aq, w16, w, bias, out);
    } else {
        gemm_kernel<false><<<grid, 256, 0, stream>>>(aq, nullptr, w, bias, out);
    }
}
